// HiddenStateAdaptation_69355131896633
// MI455X (gfx1250) — matrix operand order repaired and checked by reading, not compiled, not run on hardware
//
#include <hip/hip_runtime.h>
#include <hip/hip_bf16.h>

typedef _Float16 v16h __attribute__((ext_vector_type(16)));
typedef _Float16 v8h  __attribute__((ext_vector_type(8)));
typedef float    v8f  __attribute__((ext_vector_type(8)));

#define B_   8
#define C_   64
#define H_   256
#define W_   256
#define HW_  (H_ * W_)
#define NPIX_ (B_ * HW_)

// ---------------------------------------------------------------------------
// Kernel A: fused  conv(3->64,3x3,pad1) -> ReLU -> conv(64->9,1x1)  via WMMA
// implicit GEMM. Block = 8 waves = 128 consecutive pixels of one image row.
// All operands staged through LDS as f16; every load is unconditional
// (clamped index + select-zero) so the compiler can clause them.
// ---------------------------------------------------------------------------
__global__ __launch_bounds__(256) void kproj_wmma(const float* __restrict__ lr,
                                                  const float* __restrict__ w1f,
                                                  const float* __restrict__ b1,
                                                  const float* __restrict__ w2f,
                                                  const float* __restrict__ b2,
                                                  float* __restrict__ kern)
{
    // lr halo tile: [cin*3+iy][x0-1 .. x0+128] (130 used, stride 144)
    __shared__ alignas(16) _Float16 lrT[9 * 144];
    // w1 padded [64][27->32], stride 40 halves (80B: 16B-aligned reads,
    // conflict-free across 16 lanes)
    __shared__ alignas(16) _Float16 w1T[64 * 40];
    // w2 padded [9->16][64], stride 72 halves (144B: aligned + conflict-free)
    __shared__ alignas(16) _Float16 w2T[16 * 72];
    // per-wave relu(conv1) repack tile [pixel 16][channel 64], stride 72
    __shared__ alignas(16) _Float16 tile[8][16 * 72];

    const int tid  = threadIdx.x;
    const int lane = tid & 31;
    const int wave = tid >> 5;
    const int col  = lane & 15;   // N index (pixel)
    const int half = lane >> 4;   // lane half selects K/M sub-block

    const int p0 = blockIdx.x * 128;          // exact coverage, single row
    const int b  = p0 >> 16;
    const int y  = (p0 & (HW_ - 1)) >> 8;
    const int x0 = p0 & (W_ - 1);             // 0 or 128

    // ---- cooperative stage: lr halo (branchless: clamp + select) ----
    for (int idx = tid; idx < 9 * 130; idx += 256) {
        const int rrow = idx / 130;           // cin*3 + iy
        const int xx   = idx - rrow * 130;
        const int cin  = rrow / 3;
        const int iy   = rrow - cin * 3;
        const int yy   = y + iy - 1;
        const int gx   = x0 + xx - 1;
        const int yc   = yy < 0 ? 0 : (yy > H_ - 1 ? H_ - 1 : yy);
        const int gc   = gx < 0 ? 0 : (gx > W_ - 1 ? W_ - 1 : gx);
        const float v  = lr[((b * 3 + cin) * H_ + yc) * W_ + gc];
        const bool ok  = ((unsigned)yy < (unsigned)H_) && ((unsigned)gx < (unsigned)W_);
        lrT[rrow * 144 + xx] = (_Float16)(ok ? v : 0.f);
    }
    // ---- cooperative stage: w1 zero-padded to K=32 ----
    for (int idx = tid; idx < 64 * 32; idx += 256) {
        const int M = idx >> 5;
        const int K = idx & 31;
        const float v = w1f[M * 27 + (K < 27 ? K : 26)];
        w1T[M * 40 + K] = (_Float16)(K < 27 ? v : 0.f);
    }
    // ---- cooperative stage: w2 zero-padded to M=16 ----
    for (int idx = tid; idx < 16 * 64; idx += 256) {
        const int M = idx >> 6;
        const int K = idx & 63;
        const float v = w2f[(M < 9 ? M : 8) * 64 + K];
        w2T[M * 72 + K] = (_Float16)(M < 9 ? v : 0.f);
    }
    __syncthreads();

    const int xl = wave * 16 + col;           // 0..127 within block row

    // ---- B operand of GEMM1: patch column, K = cin*9+iy*3+jx = row*3+jx ----
    // B as built here (contiguous k order): lane = column N, element e -> K = half*16+e.
    // LDS offset for K: (K/3)*144 + K%3; select between two constants on half.
    v16h bm1;
#pragma unroll
    for (int e = 0; e < 16; ++e) {
        const int k0  = e;                                  // half==0 K
        const int k1e = 16 + e;                             // half==1 K
        const int k1  = k1e < 27 ? k1e : 26;                // clamped
        const int o0  = (k0 / 3) * 144 + (k0 % 3);
        const int o1  = (k1 / 3) * 144 + (k1 % 3);
        const _Float16 v = lrT[(half ? o1 : o0) + xl];
        bm1[e] = (!half || k1e < 27) ? v : (_Float16)0.f;   // folds for e<11
    }

    // ---- GEMM1: four 16x16x32 WMMAs over the 64 output channels ----
    // A as built here (contiguous k, same as B): lane = M; element e -> K=half*16+e.
    v8f acc[4];
#pragma unroll
    for (int mt = 0; mt < 4; ++mt) {
        const _Float16* wrow = &w1T[(mt * 16 + col) * 40 + half * 16];
        const v8h a_lo = *(const v8h*)(wrow);
        const v8h a_hi = *(const v8h*)(wrow + 8);
        v16h am;
#pragma unroll
        for (int e = 0; e < 8; ++e) { am[e] = a_lo[e]; am[e + 8] = a_hi[e]; }
        v8f z = {};
        acc[mt] = __builtin_amdgcn_wmma_f32_16x16x32_f16(
            false, am, false, bm1, (short)0, z, false, false);
    }

    // ---- bias + ReLU, repack C-layout -> LDS [pixel][channel] f16 ----
    // C layout: VGPR r -> M = r + half*8, N = col.
#pragma unroll
    for (int mt = 0; mt < 4; ++mt) {
        const float4 bA = *(const float4*)&b1[mt * 16 + half * 8];
        const float4 bB = *(const float4*)&b1[mt * 16 + half * 8 + 4];
        v8h h8;
        h8[0] = (_Float16)fmaxf(acc[mt][0] + bA.x, 0.f);
        h8[1] = (_Float16)fmaxf(acc[mt][1] + bA.y, 0.f);
        h8[2] = (_Float16)fmaxf(acc[mt][2] + bA.z, 0.f);
        h8[3] = (_Float16)fmaxf(acc[mt][3] + bA.w, 0.f);
        h8[4] = (_Float16)fmaxf(acc[mt][4] + bB.x, 0.f);
        h8[5] = (_Float16)fmaxf(acc[mt][5] + bB.y, 0.f);
        h8[6] = (_Float16)fmaxf(acc[mt][6] + bB.z, 0.f);
        h8[7] = (_Float16)fmaxf(acc[mt][7] + bB.w, 0.f);
        *(v8h*)&tile[wave][col * 72 + mt * 16 + half * 8] = h8;
    }

    // intra-wave LDS RAW on wave-private tile; no cross-wave barrier needed
    asm volatile("s_wait_dscnt 0" ::: "memory");

    // ---- GEMM2: [16(9) x 64] x [64 x 16], two chained 16x16x32 WMMAs ----
    v8f acc2 = {};
#pragma unroll
    for (int kc = 0; kc < 2; ++kc) {
        // B operand: channel K = kc*32 + half*16 + e, contiguous in LDS row
        const _Float16* trow = &tile[wave][col * 72 + kc * 32 + half * 16];
        const v8h blo = *(const v8h*)(trow);
        const v8h bhi = *(const v8h*)(trow + 8);
        v16h bm2;
#pragma unroll
        for (int e = 0; e < 8; ++e) { bm2[e] = blo[e]; bm2[e + 8] = bhi[e]; }

        // A operand: M = col, element e -> K = kc*32 + half*16 + e
        const _Float16* arow = &w2T[col * 72 + kc * 32 + half * 16];
        const v8h alo = *(const v8h*)(arow);
        const v8h ahi = *(const v8h*)(arow + 8);
        v16h am2;
#pragma unroll
        for (int e = 0; e < 8; ++e) { am2[e] = alo[e]; am2[e + 8] = ahi[e]; }

        acc2 = __builtin_amdgcn_wmma_f32_16x16x32_f16(
            false, am2, false, bm2, (short)0, acc2, false, false);
    }

    // ---- store kern[b, m, y, x] for taps m = r + half*8 < 9 ----
#pragma unroll
    for (int r = 0; r < 8; ++r) {
        const int m  = half * 8 + r;
        const float bb = b2[m < 9 ? m : 8];               // unconditional load
        if (m < 9)
            kern[((b * 9 + m) * H_ + y) * W_ + x0 + xl] = acc2[r] + bb;
    }
}

// ---------------------------------------------------------------------------
// Kernel B: out = h * sigmoid( sum_taps shifted(h) * kern_tap ).
// Bandwidth kernel: one float4 of output per thread, branchless halo.
// ---------------------------------------------------------------------------
__global__ __launch_bounds__(256) void adapt_apply(const float* __restrict__ h,
                                                   const float* __restrict__ kern,
                                                   float* __restrict__ out)
{
    const int t  = blockIdx.x * blockDim.x + threadIdx.x;  // one float4
    const int x0 = (t & (W_ / 4 - 1)) * 4;
    const int r1 = t >> 6;            // (b*C + c)*H + y
    const int y  = r1 & (H_ - 1);
    const int bc = r1 >> 8;           // b*C + c
    const int b  = bc >> 6;

    const int hbase = (bc * H_ + y) * W_ + x0;

    // three h rows, 6 floats each (x0-1 .. x0+4); unconditional clamped loads
    const bool lok = x0 > 0;
    const bool rok = x0 + 4 < W_;
    float row[3][6];
#pragma unroll
    for (int i = 0; i < 3; ++i) {
        const int yy  = y + i - 1;
        const bool yok = (unsigned)yy < (unsigned)H_;   // folds true for i==1
        const int ro  = (bc * H_ + (yok ? yy : y)) * W_ + x0;
        const float4 c4 = *(const float4*)&h[ro];
        const float lf  = h[ro - (lok ? 1 : 0)];
        const float rt  = h[ro + 3 + (rok ? 1 : 0)];
        row[i][0] = (yok && lok) ? lf : 0.f;
        row[i][1] = yok ? c4.x : 0.f;
        row[i][2] = yok ? c4.y : 0.f;
        row[i][3] = yok ? c4.z : 0.f;
        row[i][4] = yok ? c4.w : 0.f;
        row[i][5] = (yok && rok) ? rt : 0.f;
    }

    float sim0 = 0.f, sim1 = 0.f, sim2 = 0.f, sim3 = 0.f;
#pragma unroll
    for (int i = 0; i < 3; ++i) {
#pragma unroll
        for (int j = 0; j < 3; ++j) {
            const int m = i * 3 + j;
            const float4 km = *(const float4*)&kern[((b * 9 + m) * H_ + y) * W_ + x0];
            sim0 = fmaf(row[i][0 + j], km.x, sim0);
            sim1 = fmaf(row[i][1 + j], km.y, sim1);
            sim2 = fmaf(row[i][2 + j], km.z, sim2);
            sim3 = fmaf(row[i][3 + j], km.w, sim3);
        }
    }

    float4 o;
    o.x = row[1][1] * (1.f / (1.f + __expf(-sim0)));
    o.y = row[1][2] * (1.f / (1.f + __expf(-sim1)));
    o.z = row[1][3] * (1.f / (1.f + __expf(-sim2)));
    o.w = row[1][4] * (1.f / (1.f + __expf(-sim3)));
    *(float4*)&out[hbase] = o;
}

extern "C" void kernel_launch(void* const* d_in, const int* in_sizes, int n_in,
                              void* d_out, int out_size, void* d_ws, size_t ws_size,
                              hipStream_t stream) {
    (void)in_sizes; (void)n_in; (void)out_size; (void)ws_size;
    const float* lr = (const float*)d_in[0];   // [8,3,256,256]
    const float* h  = (const float*)d_in[1];   // [8,64,256,256]
    const float* w1 = (const float*)d_in[2];   // [64,3,3,3]
    const float* b1 = (const float*)d_in[3];   // [64]
    const float* w2 = (const float*)d_in[4];   // [9,64,1,1]
    const float* b2 = (const float*)d_in[5];   // [9]
    float* out  = (float*)d_out;               // [8,64,256,256]
    float* kern = (float*)d_ws;                // [8,9,256,256] = 18.9 MB scratch

    // 524288 pixels / 128 per block = 4096 blocks, exact coverage
    kproj_wmma<<<NPIX_ / 128, 256, 0, stream>>>(lr, w1, b1, w2, b2, kern);

    // 8*64*256*256 / 4 = 8,388,608 float4s / 256 threads = 32768 blocks
    adapt_apply<<<(B_ * C_ * HW_ / 4) / 256, 256, 0, stream>>>(h, kern, out);
}
